// GCN_18494129177104
// MI455X (gfx1250) — matrix operand order repaired and checked by reading, not compiled, not run on hardware
//
#include <hip/hip_runtime.h>

// ---------------------------------------------------------------------------
// GCN on MI455X (gfx1250): WMMA f16 GEMMs + async-LDS staging + atomic scatter
// ---------------------------------------------------------------------------

typedef __attribute__((ext_vector_type(16))) _Float16 v16h;
typedef __attribute__((ext_vector_type(8)))  float    v8f;
typedef int gcn_v4i __attribute__((vector_size(16)));

#if defined(__AMDGCN__)
#define GCN_AS1 __attribute__((address_space(1)))
#define GCN_AS3 __attribute__((address_space(3)))
#endif

// 16-byte global -> LDS copy. Prefer CDNA5 async-DMA path (ASYNCcnt-tracked),
// fall back to a synchronous b128 load + ds_store if the builtin is absent.
// Builtin signature (per hipcc diagnostics): (v4i AS1*, v4i AS3*, imm, imm).
// Pointers are round-tripped through integers to perform the addrspace cast
// (low 32 bits of a generic LDS pointer are the LDS byte offset on amdgcn).
__device__ __forceinline__ void async_cp16(const float* g, float* l) {
#if defined(__AMDGCN__) && __has_builtin(__builtin_amdgcn_global_load_async_to_lds_b128)
  __builtin_amdgcn_global_load_async_to_lds_b128(
      (GCN_AS1 gcn_v4i*)(unsigned long long)g,
      (GCN_AS3 gcn_v4i*)(unsigned int)(unsigned long long)l,
      0, 0);
#else
  *(float4*)l = *(const float4*)g;
#endif
}

__device__ __forceinline__ void async_wait_all() {
#if defined(__AMDGCN__) && __has_builtin(__builtin_amdgcn_s_wait_asynccnt)
  __builtin_amdgcn_s_wait_asynccnt(0);
#elif defined(__AMDGCN__)
  asm volatile("s_wait_asynccnt 0x0" ::: "memory");
#endif
}

// ---------------------------------------------------------------------------
// Weight pre-swizzle: f32 [K x 128] -> f16 fragments laid out exactly as the
// WMMA 16x16x32 B-matrix uses them here (contiguous k, same as A): [K/32 chunks][8 col-tiles][32 lanes][16 h]
// lane L holds column n = L%16, K-base = 16*(L/16); halves j=0..15 -> K-base+j.
// ---------------------------------------------------------------------------
__global__ void pack_w_f16(const float* __restrict__ W,
                           unsigned short* __restrict__ Wh, int total) {
  int idx = blockIdx.x * 256 + threadIdx.x;
  if (idx >= total) return;
  int j     = idx & 15;
  int lane  = (idx >> 4) & 31;
  int tile  = (idx >> 9) & 7;
  int chunk = idx >> 12;
  int krow = chunk * 32 + (lane >> 4) * 16 + j;
  int coln = tile * 16 + (lane & 15);
  _Float16 h = (_Float16)W[krow * 128 + coln];
  unsigned short us;
  __builtin_memcpy(&us, &h, 2);
  Wh[idx] = us;
}

// ---------------------------------------------------------------------------
// WMMA GEMM: C[N x 128] = concat_K(A0,A1,A2)[N x NSRC*128] @ W + bias (opt relu)
// Block = 256 thr (8 waves). Block computes 16 rows x 128 cols; wave w owns the
// 16x16 tile at columns [16w,16w+16). A tile staged f32 in LDS (async DMA),
// converted to f16 at fragment build. f32 accumulation via v_wmma.
// ---------------------------------------------------------------------------
template <int NSRC>
__global__ void __launch_bounds__(256)
gemm16_wmma(const float* __restrict__ A0, const float* __restrict__ A1,
            const float* __restrict__ A2,
            const unsigned short* __restrict__ Wh,
            const float* __restrict__ bias,
            float* __restrict__ C, int do_relu) {
  constexpr int KTOT = NSRC * 128;
  constexpr int LDA  = KTOT + 4;           // +4 dwords: conflict-free ds_b128
  __shared__ float As[16 * LDA];

  const int tid  = threadIdx.x;
  const int wave = tid >> 5;
  const int lane = tid & 31;
  const long long row0 = (long long)blockIdx.x * 16;

  const float* srcs[3] = {A0, A1, A2};

  // Stage the 16 x KTOT f32 A-tile into LDS with async b128 copies.
  constexpr int NOPS = (16 * KTOT) / (4 * 256);
  #pragma unroll
  for (int o = 0; o < NOPS; ++o) {
    int d = (o * 256 + tid) * 4;           // dword index in tile
    int r = d / KTOT;
    int c = d % KTOT;
    const float* g = srcs[c >> 7] + (row0 + r) * 128 + (c & 127);
    async_cp16(g, &As[r * LDA + c]);
  }
  async_wait_all();
  __syncthreads();

  const int m    = lane & 15;              // A row / output column within tile
  const int half = lane >> 4;
  const int col  = wave * 16 + m;          // output column this lane stores

  v8f acc = {};
  #pragma unroll
  for (int ch = 0; ch < KTOT / 32; ++ch) {
    // B fragment: pre-swizzled, two coalesced 16B loads per lane.
    const unsigned short* wp = Wh + (((ch * 8 + wave) * 32 + lane) << 4);
    union { v16h v; uint4 q[2]; } b;
    b.q[0] = *(const uint4*)(wp);
    b.q[1] = *(const uint4*)(wp + 8);

    // A fragment: lane L row m=L%16; halves 0-15 <- K=32ch+16*half+[0,16).
    // Four ds_b128 loads, cvt to f16.
    const float* ap = &As[m * LDA + ch * 32 + half * 16];
    float4 f0 = *(const float4*)(ap);
    float4 f1 = *(const float4*)(ap + 4);
    float4 f2 = *(const float4*)(ap + 8);
    float4 f3 = *(const float4*)(ap + 12);
    v16h a;
    a[0]  = (_Float16)f0.x; a[1]  = (_Float16)f0.y;
    a[2]  = (_Float16)f0.z; a[3]  = (_Float16)f0.w;
    a[4]  = (_Float16)f1.x; a[5]  = (_Float16)f1.y;
    a[6]  = (_Float16)f1.z; a[7]  = (_Float16)f1.w;
    a[8]  = (_Float16)f2.x; a[9]  = (_Float16)f2.y;
    a[10] = (_Float16)f2.z; a[11] = (_Float16)f2.w;
    a[12] = (_Float16)f3.x; a[13] = (_Float16)f3.y;
    a[14] = (_Float16)f3.z; a[15] = (_Float16)f3.w;

    acc = __builtin_amdgcn_wmma_f32_16x16x32_f16(false, a, false, b.v,
                                                 (short)0, acc, false, false);
  }

  // D layout: lane n = L%16; VGPR r holds row 8*(L/16)+r.
  float bc = bias ? bias[col] : 0.0f;
  #pragma unroll
  for (int r = 0; r < 8; ++r) {
    int mrow = half * 8 + r;
    float v = acc[r] + bc;
    if (do_relu) v = fmaxf(v, 0.0f);
    C[(row0 + mrow) * 128 + col] = v;
  }
}

// ---------------------------------------------------------------------------
// Graph kernels
// ---------------------------------------------------------------------------
__global__ void deg_kernel(const int* __restrict__ col, float* __restrict__ deg,
                           int E) {
  int e = blockIdx.x * 256 + threadIdx.x;
  if (e < E) atomicAdd(&deg[col[e]], 1.0f);
}

__global__ void rsqrt_kernel(float* __restrict__ deg, int N) {
  int i = blockIdx.x * 256 + threadIdx.x;
  if (i < N) deg[i] = rsqrtf(deg[i] + 1.0f);  // self-loop degree
}

// agg = xw * dis^2 + bias  (self-loop term; also serves as the accumulator init)
__global__ void init_agg_kernel(const float* __restrict__ xw,
                                const float* __restrict__ dis,
                                const float* __restrict__ bias,
                                float* __restrict__ agg, int total) {
  int idx = blockIdx.x * 256 + threadIdx.x;
  if (idx >= total) return;
  int i = idx >> 7, c = idx & 127;
  float d = dis[i];
  agg[idx] = xw[idx] * d * d + bias[c];
}

// one wave per edge: agg[col] += xw[row] * dis[row]*dis[col]  (128 channels)
__global__ void __launch_bounds__(256)
edge_scatter_kernel(const int* __restrict__ row, const int* __restrict__ col,
                    const float* __restrict__ xw, const float* __restrict__ dis,
                    float* __restrict__ agg, int E) {
  int e = blockIdx.x * 8 + (threadIdx.x >> 5);
  if (e >= E) return;
  int lane = threadIdx.x & 31;
  int r = row[e], c = col[e];
  float norm = dis[r] * dis[c];
  float4 v = ((const float4*)(xw + (long long)r * 128))[lane];
  float* dst = agg + (long long)c * 128 + lane * 4;
  atomicAdd(dst + 0, v.x * norm);
  atomicAdd(dst + 1, v.y * norm);
  atomicAdd(dst + 2, v.z * norm);
  atomicAdd(dst + 3, v.w * norm);
}

__global__ void relu_kernel(const float* __restrict__ in,
                            float* __restrict__ out, int total) {
  int idx = blockIdx.x * 256 + threadIdx.x;
  if (idx < total) out[idx] = fmaxf(in[idx], 0.0f);
}

// out[i] = sigmoid(dot(h1[i,:128], Wl2) + bl2) ; one wave per node
__global__ void __launch_bounds__(256)
final_kernel(const float* __restrict__ h1, const float* __restrict__ Wl2,
             const float* __restrict__ bl2, float* __restrict__ out, int N) {
  int i = blockIdx.x * 8 + (threadIdx.x >> 5);
  if (i >= N) return;
  int lane = threadIdx.x & 31;
  float4 h = ((const float4*)(h1 + (long long)i * 128))[lane];
  float4 w = ((const float4*)Wl2)[lane];
  float s = h.x * w.x + h.y * w.y + h.z * w.z + h.w * w.w;
  #pragma unroll
  for (int off = 16; off; off >>= 1) s += __shfl_down(s, off, 32);
  if (lane == 0) out[i] = 1.0f / (1.0f + expf(-(s + bl2[0])));
}

static inline int cdiv(int a, int b) { return (a + b - 1) / b; }

extern "C" void kernel_launch(void* const* d_in, const int* in_sizes, int n_in,
                              void* d_out, int out_size, void* d_ws,
                              size_t ws_size, hipStream_t stream) {
  const float* x   = (const float*)d_in[0];
  const int*   ei  = (const int*)d_in[1];
  const float* W1  = (const float*)d_in[2];
  const float* b1  = (const float*)d_in[3];
  const float* W2  = (const float*)d_in[4];
  const float* b2  = (const float*)d_in[5];
  const float* Wl1 = (const float*)d_in[6];
  const float* bl1 = (const float*)d_in[7];
  const float* Wl2 = (const float*)d_in[8];
  const float* bl2 = (const float*)d_in[9];

  const int N = in_sizes[0] / 128;
  const int E = in_sizes[1] / 2;
  const int* row = ei;       // edge_index[0] = source
  const int* col = ei + E;   // edge_index[1] = target

  // workspace carve-up (~205 MB)
  char* ws = (char*)d_ws;
  auto carve = [&](size_t bytes) {
    char* p = ws;
    ws += (bytes + 255) & ~(size_t)255;
    return p;
  };
  float* dis  = (float*)carve((size_t)N * 4);         // deg, then rsqrt(deg)
  float* bufA = (float*)carve((size_t)N * 128 * 4);   // xw / h1
  float* bufB = (float*)carve((size_t)N * 128 * 4);   // agg accumulator
  float* x1   = (float*)carve((size_t)N * 128 * 4);
  float* x2   = (float*)carve((size_t)N * 128 * 4);
  unsigned short* Wh1  = (unsigned short*)carve(128 * 128 * 2);
  unsigned short* Wh2  = (unsigned short*)carve(128 * 128 * 2);
  unsigned short* Whl1 = (unsigned short*)carve(384 * 128 * 2);

  const int TOT = N * 128;

  // degrees
  (void)hipMemsetAsync(dis, 0, (size_t)N * 4, stream);
  // weight pre-swizzle (independent of degree work)
  pack_w_f16<<<cdiv(128 * 128, 256), 256, 0, stream>>>(W1, Wh1, 128 * 128);
  pack_w_f16<<<cdiv(128 * 128, 256), 256, 0, stream>>>(W2, Wh2, 128 * 128);
  pack_w_f16<<<cdiv(384 * 128, 256), 256, 0, stream>>>(Wl1, Whl1, 384 * 128);
  deg_kernel<<<cdiv(E, 256), 256, 0, stream>>>(col, dis, E);
  rsqrt_kernel<<<cdiv(N, 256), 256, 0, stream>>>(dis, N);

  // ---- layer 1: x1 = relu(GCNConv(x)) ----
  gemm16_wmma<1><<<N / 16, 256, 0, stream>>>(x, nullptr, nullptr, Wh1, nullptr,
                                             bufA, 0);
  init_agg_kernel<<<cdiv(TOT, 256), 256, 0, stream>>>(bufA, dis, b1, bufB, TOT);
  edge_scatter_kernel<<<cdiv(E, 8), 256, 0, stream>>>(row, col, bufA, dis,
                                                      bufB, E);
  relu_kernel<<<cdiv(TOT, 256), 256, 0, stream>>>(bufB, x1, TOT);

  // ---- layer 2: x2 = relu(GCNConv(x1)) ----
  gemm16_wmma<1><<<N / 16, 256, 0, stream>>>(x1, nullptr, nullptr, Wh2, nullptr,
                                             bufA, 0);
  init_agg_kernel<<<cdiv(TOT, 256), 256, 0, stream>>>(bufA, dis, b2, bufB, TOT);
  edge_scatter_kernel<<<cdiv(E, 8), 256, 0, stream>>>(row, col, bufA, dis,
                                                      bufB, E);
  relu_kernel<<<cdiv(TOT, 256), 256, 0, stream>>>(bufB, x2, TOT);

  // ---- MLP head: h1 = relu([x,x1,x2] @ Wl1 + bl1); out = sigmoid(h1@Wl2+bl2)
  gemm16_wmma<3><<<N / 16, 256, 0, stream>>>(x, x1, x2, Whl1, bl1, bufA, 1);
  final_kernel<<<cdiv(N, 8), 256, 0, stream>>>(bufA, Wl2, bl2, (float*)d_out, N);
}
